// SwitchLinear_7404523618415
// MI455X (gfx1250) — matrix operand order repaired and checked by reading, not compiled, not run on hardware
//
#include <hip/hip_runtime.h>
#include <hip/hip_bf16.h>

// ---------------------------------------------------------------------------
// Switch (top-1) MoE layer for MI455X (gfx1250, wave32, WMMA).
// B=4, S=2048 -> T=8192 tokens, D_IN=D_OUT=1024, E=8.
// ---------------------------------------------------------------------------

#define T_TOK   8192
#define D_IN    1024
#define D_OUT   1024
#define NEXP    8

// workspace layout (bytes)
#define WS_WEBF   0u                               // 8*1024*1024 bf16 = 16 MB
#define WS_GATES  (16u*1024u*1024u)                // 8192*8 f32 = 256 KB
#define WS_TOP1   (WS_GATES + 8192u*8u*4u)         // 8192 int
#define WS_BUCKET (WS_TOP1 + 8192u*4u)             // 8192 int
#define WS_META   (WS_BUCKET + 8192u*4u)           // counts[8], offsets[8], cursors[8]

typedef __attribute__((ext_vector_type(16))) __bf16 v16bf;
typedef __attribute__((ext_vector_type(8)))  float  v8f;

union FragBF {
    v16bf v;
    uint4 u[2];
};

#if defined(__has_builtin)
#  if __has_builtin(__builtin_amdgcn_sched_barrier)
#    define SCHED_BARRIER() __builtin_amdgcn_sched_barrier(0)
#  endif
#endif
#ifndef SCHED_BARRIER
#  define SCHED_BARRIER() asm volatile("" ::: "memory")
#endif

__device__ __forceinline__ unsigned short f2bf(float f) {
    unsigned int u = __float_as_uint(f);
    unsigned int r = u + 0x7FFFu + ((u >> 16) & 1u);   // round-to-nearest-even
    return (unsigned short)(r >> 16);
}
__device__ __forceinline__ unsigned int pack2(float a, float b) {
    return (unsigned int)f2bf(a) | ((unsigned int)f2bf(b) << 16);
}

// --------------------------- kernel 0: zero meta ---------------------------
__global__ void k_init(int* meta) {
    if (threadIdx.x < 24) meta[threadIdx.x] = 0;   // counts[8], offsets[8], cursors[8]
}

// ---------------------- kernel 1: convert We fp32->bf16 --------------------
__global__ __launch_bounds__(256) void k_cvt_we(const float* __restrict__ We,
                                               unsigned short* __restrict__ We_bf) {
    int i = (blockIdx.x * 256 + threadIdx.x) * 8;          // 8 elems / thread
    float4 a0 = *(const float4*)(We + i);
    float4 a1 = *(const float4*)(We + i + 4);
    uint4 pk;
    pk.x = pack2(a0.x, a0.y); pk.y = pack2(a0.z, a0.w);
    pk.z = pack2(a1.x, a1.y); pk.w = pack2(a1.z, a1.w);
    *(uint4*)(We_bf + i) = pk;
}

// ------------------- kernel 2: gating (one wave per token) -----------------
__global__ __launch_bounds__(256) void k_gate(const float* __restrict__ x,
                                              const float* __restrict__ Wg,
                                              const float* __restrict__ bg,
                                              float* __restrict__ gates,
                                              int*   __restrict__ top1,
                                              int*   __restrict__ counts) {
    __shared__ float sWg[NEXP * D_IN];            // 32 KB
    int tid = threadIdx.x;
    for (int i = tid; i < NEXP * D_IN; i += 256) sWg[i] = Wg[i];
    __syncthreads();

    int wave = tid >> 5, lane = tid & 31;
    int t = blockIdx.x * 8 + wave;                // 8 tokens per block
    const float* xr = x + t * D_IN;

    float acc[NEXP];
#pragma unroll
    for (int e = 0; e < NEXP; ++e) acc[e] = 0.f;
    for (int d = lane; d < D_IN; d += 32) {
        float xv = xr[d];
#pragma unroll
        for (int e = 0; e < NEXP; ++e) acc[e] += xv * sWg[e * D_IN + d];
    }
#pragma unroll
    for (int off = 16; off > 0; off >>= 1) {
#pragma unroll
        for (int e = 0; e < NEXP; ++e) acc[e] += __shfl_xor(acc[e], off, 32);
    }
    if (lane == 0) {
        float l[NEXP];
        float m = -1e30f;
#pragma unroll
        for (int e = 0; e < NEXP; ++e) { l[e] = acc[e] + bg[e]; m = fmaxf(m, l[e]); }
        float s = 0.f;
#pragma unroll
        for (int e = 0; e < NEXP; ++e) { l[e] = __expf(l[e] - m); s += l[e]; }
        float inv = 1.f / s;
        int best = 0;
#pragma unroll
        for (int e = 0; e < NEXP; ++e) {
            float g = l[e] * inv;
            gates[t * NEXP + e] = g;
            if (e > 0 && l[e] > l[best]) best = e;   // first-max tie rule
        }
        top1[t] = best;
        atomicAdd(&counts[best], 1);
    }
}

// --------------------- kernel 3: exclusive prefix offsets ------------------
__global__ void k_offsets(const int* __restrict__ counts, int* __restrict__ offsets) {
    if (threadIdx.x == 0) {
        int run = 0;
        for (int e = 0; e < NEXP; ++e) { offsets[e] = run; run += counts[e]; }
    }
}

// ----------------------- kernel 4: scatter into buckets --------------------
__global__ __launch_bounds__(256) void k_scatter(const int* __restrict__ top1,
                                                 const int* __restrict__ offsets,
                                                 int* __restrict__ cursors,
                                                 int* __restrict__ bucket) {
    int t = blockIdx.x * 256 + threadIdx.x;
    int e = top1[t];
    int slot = atomicAdd(&cursors[e], 1);
    bucket[offsets[e] + slot] = t;
}

// ------------------- kernel 5: aux loss (deterministic) --------------------
__global__ __launch_bounds__(256) void k_aux(const float* __restrict__ gates,
                                             float* __restrict__ aux_out) {
    __shared__ float red[NEXP][32];
    int tid = threadIdx.x;
    int e = tid & 7, stripe = tid >> 3;           // 32 stripes per expert
    float local = 0.f;
    for (int t = stripe; t < T_TOK; t += 32) local += gates[t * NEXP + e];
    red[e][stripe] = local;
    __syncthreads();
    if (tid == 0) {
        float aux = 0.f;
        for (int ee = 0; ee < NEXP; ++ee) {
            float s = 0.f;
            for (int j = 0; j < 32; ++j) s += red[ee][j];   // fixed order
            float v = (s / (float)T_TOK) * (float)NEXP;
            aux += v * v;
        }
        aux_out[0] = aux / (float)NEXP;
    }
}

// ---------------- kernel 6: bucketed expert GEMM with WMMA -----------------
// grid.x = 520 (max row tiles), grid.y = 2 (512 columns each), 256 threads.
// Each wave owns 64 output columns = four 16x16 accumulators. The K loop is
// a modulo-3 register pipeline (prefetch distance 2) with an explicit
// __builtin_amdgcn_sched_barrier(0) between the prefetch loads of stage it+2
// and the WMMAs of stage it, so the backend scheduler cannot sink the loads
// back to their uses (which previously produced s_wait_loadcnt 0x0 stalls).
#define APAD 8
#define ASTRIDE (D_IN + APAD)                     // 1032 shorts per row
#define KSTEP 32
#define KITER (D_IN / KSTEP)                      // 32

__global__ __launch_bounds__(256) void k_moe_gemm(const float* __restrict__ x,
                                                  const unsigned short* __restrict__ We_bf,
                                                  const float* __restrict__ be,
                                                  const int* __restrict__ bucket,
                                                  const int* __restrict__ counts,
                                                  float* __restrict__ out) {
    // ---- map flat row-tile id -> (expert, local tile) ----
    int tileId = blockIdx.x;
    int e = -1, local = tileId, base = 0;
#pragma unroll
    for (int ee = 0; ee < NEXP; ++ee) {
        int c = counts[ee];
        int tiles = (c + 15) >> 4;
        if (e < 0) {
            if (local < tiles) { e = ee; }
            else { local -= tiles; base += c; }
        }
    }
    if (e < 0) return;
    int cnt      = counts[e];
    int rowStart = local * 16;
    int nrows    = min(16, cnt - rowStart);

    __shared__ int sTok[16];
    __shared__ __align__(16) unsigned short sA[16 * ASTRIDE];   // ~32.3 KB

    int tid = threadIdx.x;
    if (tid < 16) sTok[tid] = (tid < nrows) ? bucket[base + rowStart + tid] : -1;
    __syncthreads();

    // ---- stage A: gather 16 token rows, convert fp32 -> bf16 into LDS ----
    {
        int r = tid & 15, c = tid >> 4;                         // 64 dims per (r,c)
        int tok = sTok[r];
        const float4* xr = (const float4*)(x + (tok >= 0 ? tok : 0) * D_IN);
#pragma unroll
        for (int j = 0; j < 8; ++j) {
            int d = c * 64 + j * 8;
            float4 a0 = xr[d / 4];
            float4 a1 = xr[d / 4 + 1];
            uint4 pk;
            pk.x = pack2(a0.x, a0.y); pk.y = pack2(a0.z, a0.w);
            pk.z = pack2(a1.x, a1.y); pk.w = pack2(a1.z, a1.w);
            *(uint4*)&sA[r * ASTRIDE + d] = pk;
        }
    }
    __syncthreads();

    // ---- WMMA: each wave owns 64 output columns (four 16x16 tiles) ----
    int wave = tid >> 5, lane = tid & 31;
    int n0   = blockIdx.y * 512 + wave * 64;
    int arow = lane & 15, half = lane >> 4;
    int ncol = lane & 15;

    const unsigned short* Wb = We_bf + (size_t)e * D_OUT * D_IN;
    const unsigned short* bp[4];
#pragma unroll
    for (int q = 0; q < 4; ++q)
        bp[q] = Wb + (size_t)(n0 + q * 16 + ncol) * D_IN + half * 8;

    // A operand: lanes 0-15 hold K 0-7 & 16-23, lanes 16-31 hold K 8-15 & 24-31
    const unsigned short* aRow = &sA[arow * ASTRIDE + half * 8];

    v8f acc[4];
#pragma unroll
    for (int q = 0; q < 4; ++q) acc[q] = (v8f){};

    FragBF aP[3];
    FragBF bP[3][4];

#define LOAD_STAGE(st, k0)                                                     \
    do {                                                                       \
        aP[st].u[0] = *(const uint4*)(aRow + (k0));                            \
        aP[st].u[1] = *(const uint4*)(aRow + (k0) + 16);                       \
        _Pragma("unroll")                                                      \
        for (int q = 0; q < 4; ++q) {                                          \
            bP[st][q].u[0] = *(const uint4*)(bp[q] + (k0));                    \
            bP[st][q].u[1] = *(const uint4*)(bp[q] + (k0) + 16);               \
        }                                                                      \
    } while (0)

    LOAD_STAGE(0, 0);
    LOAD_STAGE(1, KSTEP);

#pragma unroll
    for (int it = 0; it < KITER; ++it) {
        const int cur = it % 3;
        if (it + 2 < KITER) {
            const int pf = (it + 2) % 3;
            LOAD_STAGE(pf, (it + 2) * KSTEP);      // issue loads for stage it+2
        }
        SCHED_BARRIER();                           // pin loads above the WMMAs
#pragma unroll
        for (int q = 0; q < 4; ++q) {
            acc[q] = __builtin_amdgcn_wmma_f32_16x16x32_bf16(false, aP[cur].v,
                                                             false, bP[cur][q].v,
                                                             (short)0, acc[q],
                                                             false, false);
        }
        SCHED_BARRIER();                           // keep stages from merging
    }
#undef LOAD_STAGE

    // ---- epilogue: add bias, scatter rows to their tokens ----
    float bias[4];
#pragma unroll
    for (int q = 0; q < 4; ++q) bias[q] = be[e * D_OUT + n0 + q * 16 + ncol];

#pragma unroll
    for (int j = 0; j < 8; ++j) {
        int m = j + 8 * half;            // C layout: VGPR j -> M=j (lanes 0-15), M=j+8 (16-31)
        if (m < nrows) {
            size_t rowOff = (size_t)sTok[m] * D_OUT + n0 + ncol;
#pragma unroll
            for (int q = 0; q < 4; ++q)
                out[rowOff + q * 16] = acc[q][j] + bias[q];
        }
    }
}

// ---------------------------------------------------------------------------
extern "C" void kernel_launch(void* const* d_in, const int* in_sizes, int n_in,
                              void* d_out, int out_size, void* d_ws, size_t ws_size,
                              hipStream_t stream) {
    const float* x  = (const float*)d_in[0];
    const float* We = (const float*)d_in[1];
    const float* be = (const float*)d_in[2];
    const float* Wg = (const float*)d_in[3];
    const float* bg = (const float*)d_in[4];

    float* out = (float*)d_out;                       // [8192*1024] then aux at end
    float* aux = out + (size_t)T_TOK * D_OUT;

    char* ws = (char*)d_ws;
    unsigned short* We_bf = (unsigned short*)(ws + WS_WEBF);
    float* gates   = (float*)(ws + WS_GATES);
    int*   top1    = (int*)  (ws + WS_TOP1);
    int*   bucket  = (int*)  (ws + WS_BUCKET);
    int*   meta    = (int*)  (ws + WS_META);
    int*   counts  = meta;
    int*   offsets = meta + 8;
    int*   cursors = meta + 16;

    k_init<<<1, 32, 0, stream>>>(meta);
    k_cvt_we<<<(NEXP * D_OUT * D_IN) / (256 * 8), 256, 0, stream>>>(We, We_bf);
    k_gate<<<T_TOK / 8, 256, 0, stream>>>(x, Wg, bg, gates, top1, counts);
    k_offsets<<<1, 32, 0, stream>>>(counts, offsets);
    k_scatter<<<T_TOK / 256, 256, 0, stream>>>(top1, offsets, cursors, bucket);
    k_aux<<<1, 256, 0, stream>>>(gates, aux);
    k_moe_gemm<<<dim3(520, 2), 256, 0, stream>>>(x, We_bf, be, bucket, counts, out);
}
